// CLIPVisionTower_46566035423610
// MI455X (gfx1250) — matrix operand order repaired and checked by reading, not compiled, not run on hardware
//
#include <hip/hip_runtime.h>

// ---------------- problem constants ----------------
#define B_ 16
#define T_ 577
#define TP_ 608        // T padded up to multiple of 32 (key dim padding for vT)
#define E_ 1024
#define H_ 16
#define D_ 64
#define SCALE_Q 0.125f // D^-0.5 = 64^-0.5

typedef __attribute__((ext_vector_type(16))) __bf16 v16bf;
typedef __attribute__((ext_vector_type(8)))  float  v8f;

union BFrag { v16bf v; unsigned short u[16]; };
struct alignas(16) US8 { unsigned short u[8]; };

__device__ __forceinline__ unsigned short f2bf(float f) {
  unsigned u = __builtin_bit_cast(unsigned, f);
  u += 0x7FFFu + ((u >> 16) & 1u);          // round-to-nearest-even
  return (unsigned short)(u >> 16);
}
__device__ __forceinline__ float bf2f(unsigned short h) {
  unsigned u = (unsigned)h << 16;
  return __builtin_bit_cast(float, u);
}

// ---------------- elementwise kernels ----------------
__global__ void k_cvt_bf16(const float* __restrict__ in, unsigned short* __restrict__ out, int n) {
  for (int i = blockIdx.x * blockDim.x + threadIdx.x; i < n; i += gridDim.x * blockDim.x)
    out[i] = f2bf(in[i]);
}
__global__ void k_logsize(const float* __restrict__ sz, float* __restrict__ ls, int n) {
  for (int i = blockIdx.x * blockDim.x + threadIdx.x; i < n; i += gridDim.x * blockDim.x)
    ls[i] = __logf(sz[i]);
}
__global__ void k_zero(float* __restrict__ p, int n) {
  for (int i = blockIdx.x * blockDim.x + threadIdx.x; i < n; i += gridDim.x * blockDim.x)
    p[i] = 0.0f;
}
// mean over heads of K: out[b,t,d] = (1/16) sum_h k[b,t,h*64+d]
__global__ void k_mean_keys(const unsigned short* __restrict__ kb, float* __restrict__ out, int n) {
  for (int i = blockIdx.x * blockDim.x + threadIdx.x; i < n; i += gridDim.x * blockDim.x) {
    int d = i & (D_ - 1);
    int bt = i >> 6;
    const unsigned short* p = kb + (size_t)bt * E_ + d;
    float s = 0.0f;
#pragma unroll
    for (int h = 0; h < H_; ++h) s += bf2f(p[h * D_]);
    out[i] = s * (1.0f / (float)H_);
  }
}

// ---------------- WMMA GEMM: C[M,N] = (A[M,K] @ W[N,K]^T + bias) * scale ----
// A, W row-major bf16; bias f32.
// OM==0: f32 row-major out; OM==1: bf16 row-major out;
// OM==2: bf16 out in head-transposed layout vT[(b*E + col)*TP_ + t] (row = b*577+t)
// block = 128 threads (4 waves), each wave does a 16x64 tile; grid (N/256, M/16)
template <int OM>
__global__ __launch_bounds__(128) void k_gemm(const unsigned short* __restrict__ A,
                                              const unsigned short* __restrict__ W,
                                              const float* __restrict__ bias, float scale,
                                              void* __restrict__ outp, int N, int K) {
  const int lane = threadIdx.x & 31, wave = threadIdx.x >> 5;
  const int r = lane & 15, hi = lane >> 4;
  const int m0 = blockIdx.y << 4;
  const int n0 = blockIdx.x * 256 + wave * 64;

  const v8f z8 = {0, 0, 0, 0, 0, 0, 0, 0};
  v8f c[4] = {z8, z8, z8, z8};

  const unsigned short* arow = A + (size_t)(m0 + r) * K;
  const unsigned short* wr[4];
#pragma unroll
  for (int t = 0; t < 4; ++t) wr[t] = W + (size_t)(n0 + t * 16 + r) * K;

  for (int k0 = 0; k0 < K; k0 += 32) {
    __builtin_prefetch(arow + k0 + 128, 0, 1);  // global_prefetch_b8, 256B ahead
    BFrag a;
    *(US8*)&a.u[0] = *(const US8*)(arow + k0 + hi * 8);       // K = hi*8 .. +7
    *(US8*)&a.u[8] = *(const US8*)(arow + k0 + 16 + hi * 8);  // K = 16+hi*8 .. +7
    const int kb = k0 + hi * 8;                               // B lane K = hi*8..+7, 16+hi*8..+7
#pragma unroll
    for (int t = 0; t < 4; ++t) {
      BFrag b;
      *(US8*)&b.u[0] = *(const US8*)(wr[t] + kb);
      *(US8*)&b.u[8] = *(const US8*)(wr[t] + kb + 16);
      c[t] = __builtin_amdgcn_wmma_f32_16x16x32_bf16(false, a.v, false, b.v, (short)0, c[t],
                                                     false, false);
    }
  }
#pragma unroll
  for (int t = 0; t < 4; ++t) {
    const int col = n0 + t * 16 + r;
    const float bi = bias[col];
#pragma unroll
    for (int j = 0; j < 8; ++j) {
      const float val = (c[t][j] + bi) * scale;
      const unsigned row = (unsigned)(m0 + j + 8 * hi);  // = b*577 + tt
      if constexpr (OM == 0) {
        ((float*)outp)[(size_t)row * N + col] = val;
      } else if constexpr (OM == 1) {
        ((unsigned short*)outp)[(size_t)row * N + col] = f2bf(val);
      } else {
        const unsigned bb = row / T_;
        const unsigned tt2 = row - bb * T_;
        ((unsigned short*)outp)[((size_t)bb * E_ + col) * TP_ + tt2] = f2bf(val);
      }
    }
  }
}

// ---------------- att_score = (1/256) * sum_b  q[b] @ k[b]^T  (full E dot) ----
// Heads partition E, so sum over (b,h) of per-head logits == sum over b of full-E dots.
// block = 128 (4 waves), wave does 16(t) x 16(s); grid (ceil(577/64)=10, 37)
__global__ __launch_bounds__(128) void k_score(const unsigned short* __restrict__ qb,
                                               const unsigned short* __restrict__ kbf,
                                               float* __restrict__ score) {
  const int lane = threadIdx.x & 31, wave = threadIdx.x >> 5;
  const int r = lane & 15, hi = lane >> 4;
  const int t0 = blockIdx.y << 4;
  const int s0 = blockIdx.x * 64 + wave * 16;
  const int trow = min(t0 + r, T_ - 1);
  const int srow = min(s0 + r, T_ - 1);

  v8f c = {0, 0, 0, 0, 0, 0, 0, 0};
  for (int b = 0; b < B_; ++b) {
    const unsigned short* ar = qb + (size_t)(b * T_ + trow) * E_;
    const unsigned short* br = kbf + (size_t)(b * T_ + srow) * E_;
    for (int k0 = 0; k0 < E_; k0 += 32) {
      BFrag a, bb;
      *(US8*)&a.u[0] = *(const US8*)(ar + k0 + hi * 8);
      *(US8*)&a.u[8] = *(const US8*)(ar + k0 + 16 + hi * 8);
      *(US8*)&bb.u[0] = *(const US8*)(br + k0 + hi * 8);
      *(US8*)&bb.u[8] = *(const US8*)(br + k0 + 16 + hi * 8);
      c = __builtin_amdgcn_wmma_f32_16x16x32_bf16(false, a.v, false, bb.v, (short)0, c, false,
                                                  false);
    }
  }
  const int scol = s0 + r;
  if (scol < T_) {
#pragma unroll
    for (int j = 0; j < 8; ++j) {
      const int t = t0 + j + 8 * hi;
      if (t < T_) score[(size_t)t * T_ + scol] = c[j] * (1.0f / (B_ * H_));
    }
  }
}

// ---------------- flash attention per (b, h, 16-query tile) ----------------
// One wave per block; online softmax. V comes from the s-contiguous vT layout.
__global__ __launch_bounds__(32) void k_attn(const unsigned short* __restrict__ qb,
                                             const unsigned short* __restrict__ kb,
                                             const unsigned short* __restrict__ vT,
                                             const float* __restrict__ logsz,
                                             unsigned short* __restrict__ ob) {
  const int lane = threadIdx.x & 31, r = lane & 15, hi = lane >> 4;
  const int t0 = blockIdx.x << 4, h = blockIdx.y, b = blockIdx.z;
  const int trow = min(t0 + r, T_ - 1);

  // Q as two A-frags (head dim 64 = 2 x K32)
  const unsigned short* qrow = qb + ((size_t)(b * T_ + trow) * E_ + h * D_);
  BFrag aq0, aq1;
  *(US8*)&aq0.u[0] = *(const US8*)(qrow + hi * 8);
  *(US8*)&aq0.u[8] = *(const US8*)(qrow + 16 + hi * 8);
  *(US8*)&aq1.u[0] = *(const US8*)(qrow + 32 + hi * 8);
  *(US8*)&aq1.u[8] = *(const US8*)(qrow + 48 + hi * 8);

  // per-lane vT base for the 4 d-chunks: row (b*E + h*64 + tt*16 + r), s-contiguous
  const unsigned short* vrow[4];
#pragma unroll
  for (int tt = 0; tt < 4; ++tt)
    vrow[tt] = vT + ((size_t)b * E_ + h * D_ + tt * 16 + r) * TP_;

  const v8f z8 = {0, 0, 0, 0, 0, 0, 0, 0};
  v8f co[4] = {z8, z8, z8, z8};
  float mrun[8], lrun[8];
#pragma unroll
  for (int j = 0; j < 8; ++j) { mrun[j] = -__builtin_inff(); lrun[j] = 0.0f; }

  __shared__ alignas(16) unsigned short ldsP[16 * 32];

  for (int s0 = 0; s0 < T_; s0 += 32) {
    v8f S[2];
#pragma unroll
    for (int kk = 0; kk < 2; ++kk) {
      const int scol = s0 + kk * 16 + r;
      const int sc = min(scol, T_ - 1);
      const unsigned short* krow = kb + ((size_t)(b * T_ + sc) * E_ + h * D_);
      BFrag b0, b1;
      *(US8*)&b0.u[0] = *(const US8*)(krow + hi * 8);
      *(US8*)&b0.u[8] = *(const US8*)(krow + 16 + hi * 8);
      *(US8*)&b1.u[0] = *(const US8*)(krow + 32 + hi * 8);
      *(US8*)&b1.u[8] = *(const US8*)(krow + 48 + hi * 8);
      v8f z = z8;
      z = __builtin_amdgcn_wmma_f32_16x16x32_bf16(false, aq0.v, false, b0.v, (short)0, z, false, false);
      z = __builtin_amdgcn_wmma_f32_16x16x32_bf16(false, aq1.v, false, b1.v, (short)0, z, false, false);
      const float bias = (scol < T_) ? logsz[b * T_ + scol] : -__builtin_inff();
#pragma unroll
      for (int j = 0; j < 8; ++j) z[j] += bias;
      S[kk] = z;
    }
    // ---- online softmax over these 32 keys (row reductions within 16-lane halves)
    float mloc[8];
#pragma unroll
    for (int j = 0; j < 8; ++j) mloc[j] = fmaxf(S[0][j], S[1][j]);
#pragma unroll
    for (int off = 1; off < 16; off <<= 1)
#pragma unroll
      for (int j = 0; j < 8; ++j) mloc[j] = fmaxf(mloc[j], __shfl_xor(mloc[j], off, 32));
    float alpha[8], psum[8];
#pragma unroll
    for (int j = 0; j < 8; ++j) {
      const float nm = fmaxf(mrun[j], mloc[j]);
      alpha[j] = (mrun[j] == -__builtin_inff()) ? 0.0f : __expf(mrun[j] - nm);
      mrun[j] = nm;
      S[0][j] = __expf(S[0][j] - nm);
      S[1][j] = __expf(S[1][j] - nm);
      psum[j] = S[0][j] + S[1][j];
    }
#pragma unroll
    for (int off = 1; off < 16; off <<= 1)
#pragma unroll
      for (int j = 0; j < 8; ++j) psum[j] += __shfl_xor(psum[j], off, 32);
#pragma unroll
    for (int j = 0; j < 8; ++j) lrun[j] = lrun[j] * alpha[j] + psum[j];
#pragma unroll
    for (int t = 0; t < 4; ++t)
#pragma unroll
      for (int j = 0; j < 8; ++j) co[t][j] *= alpha[j];

    // ---- P: C-layout -> LDS (row-major 16x32 bf16) -> A-layout frag
    __syncthreads();
#pragma unroll
    for (int kk = 0; kk < 2; ++kk)
#pragma unroll
      for (int j = 0; j < 8; ++j) ldsP[(j + 8 * hi) * 32 + kk * 16 + r] = f2bf(S[kk][j]);
    __syncthreads();
    BFrag ap;
    *(US8*)&ap.u[0] = *(const US8*)(&ldsP[r * 32 + hi * 8]);
    *(US8*)&ap.u[8] = *(const US8*)(&ldsP[r * 32 + 16 + hi * 8]);

    // ---- O += P(16x32) @ V(32x64): contiguous b128 loads from vT (padded keys: P==0)
#pragma unroll
    for (int tt = 0; tt < 4; ++tt) {
      BFrag bv;
      *(US8*)&bv.u[0] = *(const US8*)(vrow[tt] + s0 + hi * 8);
      *(US8*)&bv.u[8] = *(const US8*)(vrow[tt] + s0 + 16 + hi * 8);
      co[tt] = __builtin_amdgcn_wmma_f32_16x16x32_bf16(false, ap.v, false, bv.v, (short)0, co[tt],
                                                       false, false);
    }
  }
  // ---- epilogue: O /= l, store bf16 for the output projection
#pragma unroll
  for (int j = 0; j < 8; ++j) {
    const int t = t0 + j + 8 * hi;
    if (t < T_) {
      const float inv = 1.0f / lrun[j];
#pragma unroll
      for (int tt = 0; tt < 4; ++tt)
        ob[(size_t)(b * T_ + t) * E_ + h * D_ + tt * 16 + r] = f2bf(co[tt][j] * inv);
    }
  }
}

// ---------------- host-side orchestration ----------------
extern "C" void kernel_launch(void* const* d_in, const int* in_sizes, int n_in, void* d_out,
                              int out_size, void* d_ws, size_t ws_size, hipStream_t stream) {
  (void)in_sizes; (void)n_in; (void)out_size; (void)ws_size;
  const float* hs = (const float*)d_in[0];
  const float* sz = (const float*)d_in[1];
  const float* wq = (const float*)d_in[2];
  const float* bq = (const float*)d_in[3];
  const float* wk = (const float*)d_in[4];
  const float* bk = (const float*)d_in[5];
  const float* wv = (const float*)d_in[6];
  const float* bv = (const float*)d_in[7];
  const float* wo = (const float*)d_in[8];
  const float* bo = (const float*)d_in[9];
  float* out = (float*)d_out;

  const size_t NHS = (size_t)B_ * T_ * E_;    // 9,453,568
  const size_t NW = (size_t)E_ * E_;          // 1,048,576
  const size_t NVT = (size_t)B_ * E_ * TP_;   // 9,961,472 (bf16 elems)

  // workspace layout (~104 MB)
  unsigned short* hsb = (unsigned short*)d_ws;
  unsigned short* qb  = hsb + NHS;
  unsigned short* kbf = qb + NHS;
  unsigned short* obf = kbf + NHS;
  unsigned short* wqb = obf + NHS;
  unsigned short* wkb = wqb + NW;
  unsigned short* wvb = wkb + NW;
  unsigned short* wob = wvb + NW;
  float* logsz = (float*)(wob + NW);                 // 9232 f32 (36,928 B, 16B-aligned)
  unsigned short* vT = (unsigned short*)(logsz + (size_t)B_ * T_);

  // output layout: attn_output | mean_keys | att_score
  float* out_attn = out;
  float* out_meank = out + NHS;
  float* out_score = out + NHS + (size_t)B_ * T_ * D_;

  const dim3 blk(256);
  hipLaunchKernelGGL(k_cvt_bf16, dim3(2048), blk, 0, stream, hs, hsb, (int)NHS);
  hipLaunchKernelGGL(k_cvt_bf16, dim3(1024), blk, 0, stream, wq, wqb, (int)NW);
  hipLaunchKernelGGL(k_cvt_bf16, dim3(1024), blk, 0, stream, wk, wkb, (int)NW);
  hipLaunchKernelGGL(k_cvt_bf16, dim3(1024), blk, 0, stream, wv, wvb, (int)NW);
  hipLaunchKernelGGL(k_cvt_bf16, dim3(1024), blk, 0, stream, wo, wob, (int)NW);
  hipLaunchKernelGGL(k_logsize, dim3(64), blk, 0, stream, sz, logsz, B_ * T_);
  // zero vT (covers the s >= 577 padding the V-GEMM never writes)
  hipLaunchKernelGGL(k_zero, dim3(2048), blk, 0, stream, (float*)vT, (int)(NVT / 2));

  // QKV projections: M = B*T = 9232 = 577 * 16 (no M padding needed)
  const dim3 ggrid(E_ / 256, (B_ * T_) / 16);
  hipLaunchKernelGGL((k_gemm<1>), ggrid, dim3(128), 0, stream, hsb, wqb, bq, SCALE_Q, (void*)qb,
                     E_, E_);
  hipLaunchKernelGGL((k_gemm<1>), ggrid, dim3(128), 0, stream, hsb, wkb, bk, 1.0f, (void*)kbf,
                     E_, E_);
  hipLaunchKernelGGL((k_gemm<2>), ggrid, dim3(128), 0, stream, hsb, wvb, bv, 1.0f, (void*)vT,
                     E_, E_);

  // att_score as a WMMA GEMM (replaces 87M f32 atomics)
  hipLaunchKernelGGL(k_score, dim3(10, 37), dim3(128), 0, stream, qb, kbf, out_score);

  // flash attention: grid = (ceil(577/16)=37 query tiles, 16 heads, 16 batches)
  hipLaunchKernelGGL(k_attn, dim3(37, H_, B_), dim3(32), 0, stream, qb, kbf, vT, logsz, obf);

  hipLaunchKernelGGL(k_mean_keys, dim3(1024), blk, 0, stream, kbf, out_meank, B_ * T_ * D_);

  // output projection (f32 result straight into d_out)
  hipLaunchKernelGGL((k_gemm<0>), ggrid, dim3(128), 0, stream, obf, wob, bo, 1.0f,
                     (void*)out_attn, E_, E_);
}
